// MultiHeadSelfAttention_10763188044027
// MI455X (gfx1250) — matrix operand order repaired and checked by reading, not compiled, not run on hardware
//
#include <hip/hip_runtime.h>
#include <hip/hip_bf16.h>
#include <stdint.h>

// Problem constants (from reference)
#define FF   40     // sequence length
#define FP   48     // padded rows (3 x 16 M-tiles)
#define E    128    // embedding
#define DD   128    // heads * head_dim
#define HEADS 2
#define HD   64     // head dim
#define KPAD 64     // padded key-count for attn @ V

typedef __attribute__((ext_vector_type(16))) __bf16 v16bf;
typedef __attribute__((ext_vector_type(2)))  __bf16 v2bf;
typedef __attribute__((ext_vector_type(8)))  float  v8f;

union Frag {
    v16bf v;
    uint32_t u[8];
    uint4 q[2];
};

__device__ __forceinline__ uint16_t f32_to_bf16(float f) {
    union { float f; uint32_t u; } c; c.f = f;
    uint32_t r = c.u + 0x7FFFu + ((c.u >> 16) & 1u);   // RNE
    return (uint16_t)(r >> 16);
}

// Pack two f32 -> packed bf16x2 in one u32. Prefer the HW packed convert.
#if !__has_builtin(__builtin_amdgcn_cvt_pk_bf16_f32)
#warning "cvt_pk_bf16_f32 builtin NOT available: falling back to manual RNE"
#endif
__device__ __forceinline__ uint32_t pack_bf16x2(float lo, float hi) {
#if __has_builtin(__builtin_amdgcn_cvt_pk_bf16_f32)
    union { v2bf v; uint32_t u; } c;
    c.v = __builtin_amdgcn_cvt_pk_bf16_f32(lo, hi);
    return c.u;
#else
    return (uint32_t)f32_to_bf16(lo) | ((uint32_t)f32_to_bf16(hi) << 16);
#endif
}

// Store a packed bf16x2 into two row-strided LDS slots (stride `ld` elems).
__device__ __forceinline__ void store_bf16_pair(uint16_t* p, int ld, uint32_t pk) {
    p[0]  = (uint16_t)pk;          // ds_store_b16 (low half)
    p[ld] = (uint16_t)(pk >> 16);  // ds_store_b16_d16_hi (high half)
}

// A fragment (16x32 bf16, M x K), source row-major [M][ld] in LDS.
// lane: M = mbase + (lane&15); VGPR v: K = kbase + (lane>>4)*16 + v*2
__device__ __forceinline__ Frag load_a(const uint16_t* m, int ld, int mbase, int kbase, int lane) {
    Frag f;
    int mm = mbase + (lane & 15);
    int kh = (lane >> 4) * 16;
    const uint16_t* p = m + mm * ld + kbase + kh;
    f.q[0] = *(const uint4*)(p);        // K pairs 0..3
    f.q[1] = *(const uint4*)(p + 8);    // K pairs 4..7
    return f;
}

// B fragment (32x16 bf16, K x N) with source stored [N][K] row-major
// (B[k][n] = src[n][k]); 16 contiguous elements per lane.
__device__ __forceinline__ Frag load_bt(const uint16_t* m, int ld, int nbase, int kbase, int lane) {
    Frag f;
    int nn = nbase + (lane & 15);
    const uint16_t* p = m + nn * ld + kbase + (lane >> 4) * 16;
    f.q[0] = *(const uint4*)(p);
    f.q[1] = *(const uint4*)(p + 8);
    return f;
}

// Pre-packed B fragment from global memory: 8 consecutive u32 per lane.
__device__ __forceinline__ Frag load_bp(const uint32_t* packed, int frag_idx, int lane) {
    Frag f;
    const uint4* p = (const uint4*)(packed + (size_t)frag_idx * 256 + lane * 8);
    f.q[0] = p[0];
    f.q[1] = p[1];
    return f;
}

// ---- prep: pack the 4 weight matrices into WMMA B fragments in this kernel's contiguous k order ----
// frag index = ((proj*4 + ks)*8 + nt); fragment = 32 lanes x 8 u32.
// u32 element v of lane L packs W[kk][nn], W[kk+1][nn]:
//   kk = ks*32 + (L>>4)*16 + 2v,  nn = nt*16 + (L&15).
__global__ void pack_weights_kernel(const float* __restrict__ wq,
                                    const float* __restrict__ wk,
                                    const float* __restrict__ wv,
                                    const float* __restrict__ wr,
                                    uint32_t* __restrict__ wp) {
    int idx = blockIdx.x * blockDim.x + threadIdx.x;   // 4*4*8*32*8 = 32768
    if (idx >= 4 * 4 * 8 * 32 * 8) return;
    int v    = idx & 7;
    int lane = (idx >> 3) & 31;
    int nt   = (idx >> 8) & 7;
    int ks   = (idx >> 11) & 3;
    int proj = idx >> 13;
    const float* W = (proj == 0) ? wq : (proj == 1) ? wk : (proj == 2) ? wv : wr;
    int kk = ks * 32 + (lane >> 4) * 16 + 2 * v;
    int nn = nt * 16 + (lane & 15);
    wp[idx] = pack_bf16x2(W[kk * DD + nn], W[(kk + 1) * DD + nn]);
}

__global__ __launch_bounds__(256) void mhsa_kernel(const float* __restrict__ x,
                                                   const uint32_t* __restrict__ wp, // packed B frags
                                                   float* __restrict__ out) {
    __shared__ __align__(16) uint16_t xb[FP * E];      // x[b] bf16, zero-padded rows
    __shared__ __align__(16) uint16_t qs[FP * DD];     // Q bf16
    __shared__ __align__(16) uint16_t ksm[FP * DD];    // K bf16
    __shared__ __align__(16) uint16_t vt[DD * KPAD];   // V^T bf16: [dim][key], key padded to 64
    __shared__ __align__(16) uint16_t at[FP * KPAD];   // softmax(attn) bf16, K padded to 64
    __shared__ __align__(16) float    sc[FP * FP];     // scores f32
    __shared__ __align__(16) float    oa[FP * DD];     // output accumulator (starts as Res)

    const int b    = blockIdx.x;
    const int tid  = threadIdx.x;
    const int lane = tid & 31;
    const int wv   = __builtin_amdgcn_readfirstlane(tid >> 5);  // wave id as SGPR

    // ---- stage x[b] as bf16 (float2 -> packed u32), zero-pad rows [40,48) ----
    const float* xg = x + (size_t)b * (FF * E);
    uint32_t* xb32 = (uint32_t*)xb;
    for (int i = tid; i < (FF * E) / 2; i += 256) {          // 2560 u32
        float2 f2 = ((const float2*)xg)[i];
        xb32[i] = pack_bf16x2(f2.x, f2.y);
    }
    for (int i = tid; i < ((FP - FF) * E) / 2; i += 256) {   // 512 u32 of zero pad
        xb32[(FF * E) / 2 + i] = 0u;
    }
    // zero V^T pad columns (keys 48..63) for all 128 dims: 1024 u32
    uint32_t* vt32 = (uint32_t*)vt;
    for (int i = tid; i < DD * 8; i += 256) {
        int d = i >> 3;
        vt32[d * (KPAD / 2) + (FP / 2) + (i & 7)] = 0u;
    }
    __syncthreads();

    // ---- projections: Q,K,V,Res = xb @ W  (M=48, N=128, K=128) ----
    // Per projection: 3 Mtiles x 8 Ntiles = 24 tiles over 8 waves (3 each).

    // Q projection
    for (int t = wv; t < 24; t += 8) {
        int mt = t >> 3, nt = t & 7;
        v8f acc = {};
#pragma unroll
        for (int ks = 0; ks < 4; ++ks) {
            Frag a  = load_a(xb, E, mt * 16, ks * 32, lane);
            Frag bf = load_bp(wp, (0 * 4 + ks) * 8 + nt, lane);
            acc = __builtin_amdgcn_wmma_f32_16x16x32_bf16(false, a.v, false, bf.v,
                                                          (short)0, acc, false, false);
        }
        int nn = nt * 16 + (lane & 15);
        int mb = mt * 16 + 8 * (lane >> 4);
        uint16_t* qp = qs + mb * DD + nn;
#pragma unroll
        for (int r2 = 0; r2 < 4; ++r2)
            store_bf16_pair(qp + 2 * r2 * DD, DD, pack_bf16x2(acc[2 * r2], acc[2 * r2 + 1]));
    }
    // K projection
    for (int t = wv; t < 24; t += 8) {
        int mt = t >> 3, nt = t & 7;
        v8f acc = {};
#pragma unroll
        for (int ks = 0; ks < 4; ++ks) {
            Frag a  = load_a(xb, E, mt * 16, ks * 32, lane);
            Frag bf = load_bp(wp, (1 * 4 + ks) * 8 + nt, lane);
            acc = __builtin_amdgcn_wmma_f32_16x16x32_bf16(false, a.v, false, bf.v,
                                                          (short)0, acc, false, false);
        }
        int nn = nt * 16 + (lane & 15);
        int mb = mt * 16 + 8 * (lane >> 4);
        uint16_t* kp = ksm + mb * DD + nn;
#pragma unroll
        for (int r2 = 0; r2 < 4; ++r2)
            store_bf16_pair(kp + 2 * r2 * DD, DD, pack_bf16x2(acc[2 * r2], acc[2 * r2 + 1]));
    }
    // V projection -> transposed store; 8 contiguous keys per lane = one b128 store
    for (int t = wv; t < 24; t += 8) {
        int mt = t >> 3, nt = t & 7;
        v8f acc = {};
#pragma unroll
        for (int ks = 0; ks < 4; ++ks) {
            Frag a  = load_a(xb, E, mt * 16, ks * 32, lane);
            Frag bf = load_bp(wp, (2 * 4 + ks) * 8 + nt, lane);
            acc = __builtin_amdgcn_wmma_f32_16x16x32_bf16(false, a.v, false, bf.v,
                                                          (short)0, acc, false, false);
        }
        int nn = nt * 16 + (lane & 15);         // dim (row of vt)
        int mb = mt * 16 + 8 * (lane >> 4);     // first key (contiguous)
        uint4 pk;
        pk.x = pack_bf16x2(acc[0], acc[1]);
        pk.y = pack_bf16x2(acc[2], acc[3]);
        pk.z = pack_bf16x2(acc[4], acc[5]);
        pk.w = pack_bf16x2(acc[6], acc[7]);
        *(uint4*)(vt + nn * KPAD + mb) = pk;
    }
    // Res projection -> f32 accumulator
    for (int t = wv; t < 24; t += 8) {
        int mt = t >> 3, nt = t & 7;
        v8f acc = {};
#pragma unroll
        for (int ks = 0; ks < 4; ++ks) {
            Frag a  = load_a(xb, E, mt * 16, ks * 32, lane);
            Frag bf = load_bp(wp, (3 * 4 + ks) * 8 + nt, lane);
            acc = __builtin_amdgcn_wmma_f32_16x16x32_bf16(false, a.v, false, bf.v,
                                                          (short)0, acc, false, false);
        }
        int nn = nt * 16 + (lane & 15);
        int mb = mt * 16 + 8 * (lane >> 4);
#pragma unroll
        for (int r = 0; r < 8; ++r) oa[(mb + r) * DD + nn] = acc[r];
    }

    // ---- per-head attention ----
    for (int h = 0; h < HEADS; ++h) {
        const int hb = h * HD;
        __syncthreads();   // Q/K/V ready (h=0) or previous head's AV done (h=1)

        // scores = Q_h @ K_h^T : M=48, N=48, K=64 ; 9 tiles
        for (int t = wv; t < 9; t += 8) {
            int mt = t / 3, nt = t % 3;
            v8f acc = {};
#pragma unroll
            for (int ks = 0; ks < 2; ++ks) {
                Frag a  = load_a (qs,  DD, mt * 16, hb + ks * 32, lane);
                Frag bf = load_bt(ksm, DD, nt * 16, hb + ks * 32, lane);
                acc = __builtin_amdgcn_wmma_f32_16x16x32_bf16(false, a.v, false, bf.v,
                                                              (short)0, acc, false, false);
            }
            int nn = nt * 16 + (lane & 15);
            int mb = mt * 16 + 8 * (lane >> 4);
#pragma unroll
            for (int r = 0; r < 8; ++r) sc[(mb + r) * FP + nn] = acc[r];
        }
        __syncthreads();

        // row softmax (valid region 40x40), write attn bf16 zero-padded to 48x64
        if (tid < FP) {
            int rb = tid * FP;
            uint32_t* ab = (uint32_t*)(at + tid * KPAD);
            if (tid < FF) {
                float mx = -3.0e38f;
                for (int c = 0; c < FF; ++c) mx = fmaxf(mx, sc[rb + c]);
                float sum = 0.f;
                for (int c = 0; c < FF; ++c) {
                    float e = __expf(sc[rb + c] - mx);
                    sc[rb + c] = e;                    // cache exp for pass 3
                    sum += e;
                }
                float inv = 1.0f / sum;
                for (int c = 0; c < FF / 2; ++c)
                    ab[c] = pack_bf16x2(sc[rb + 2 * c] * inv, sc[rb + 2 * c + 1] * inv);
                for (int c = FF / 2; c < KPAD / 2; ++c) ab[c] = 0u;
            } else {
                for (int c = 0; c < KPAD / 2; ++c) ab[c] = 0u;
            }
        }
        __syncthreads();

        // out_h = attn @ V_h : M=48, N=64, K=64 ; 12 tiles.
        // Residual/previous value rides in as the WMMA C operand (no post-add).
        // B[k=key][n=dim] = V[key][dim] = vt[dim][key] -> contiguous via load_bt.
        for (int t = wv; t < 12; t += 8) {
            int mt = t >> 2, nt = t & 3;
            int nn = hb + nt * 16 + (lane & 15);
            int mb = mt * 16 + 8 * (lane >> 4);
            v8f acc;
#pragma unroll
            for (int r = 0; r < 8; ++r) acc[r] = oa[(mb + r) * DD + nn];
#pragma unroll
            for (int ks = 0; ks < 2; ++ks) {
                Frag a  = load_a (at, KPAD, mt * 16, ks * 32, lane);
                Frag bf = load_bt(vt, KPAD, hb + nt * 16, ks * 32, lane);
                acc = __builtin_amdgcn_wmma_f32_16x16x32_bf16(false, a.v, false, bf.v,
                                                              (short)0, acc, false, false);
            }
#pragma unroll
            for (int r = 0; r < 8; ++r) oa[(mb + r) * DD + nn] = acc[r];
        }
    }
    __syncthreads();

    // ---- ReLU + vectorized coalesced store of valid rows ----
    float* og = out + (size_t)b * (FF * DD);
    const float4* oa4 = (const float4*)oa;
    float4* og4 = (float4*)og;
    for (int i = tid; i < (FF * DD) / 4; i += 256) {   // 1280 float4
        float4 v4 = oa4[i];
        v4.x = fmaxf(v4.x, 0.0f);
        v4.y = fmaxf(v4.y, 0.0f);
        v4.z = fmaxf(v4.z, 0.0f);
        v4.w = fmaxf(v4.w, 0.0f);
        og4[i] = v4;
    }
}

extern "C" void kernel_launch(void* const* d_in, const int* in_sizes, int n_in,
                              void* d_out, int out_size, void* d_ws, size_t ws_size,
                              hipStream_t stream) {
    const float* x  = (const float*)d_in[0];
    const float* wq = (const float*)d_in[1];
    const float* wk = (const float*)d_in[2];
    const float* wv = (const float*)d_in[3];
    const float* wr = (const float*)d_in[4];
    float* out = (float*)d_out;

    uint32_t* wp = (uint32_t*)d_ws;  // 32768 u32 = 128 KB packed weight fragments

    int npack = 4 * 4 * 8 * 32 * 8;
    pack_weights_kernel<<<(npack + 255) / 256, 256, 0, stream>>>(wq, wk, wv, wr, wp);

    int nbatch = in_sizes[0] / (FF * E);   // 16384
    mhsa_kernel<<<nbatch, 256, 0, stream>>>(x, wp, out);
}
